// CombinedMLPMoEModel_38912403702043
// MI455X (gfx1250) — compile-verified, matrix operand order checked by reading, not run on hardware
//
#include <hip/hip_runtime.h>
#include <stdint.h>

// ============================================================================
// CombinedMLPMoE for MI455X (gfx1250, wave32, WMMA).
// - All heavy GEMMs: v_wmma_f32_16x16x32_bf16 (bf16 operands, f32 accum).
// - Global->LDS staging uses CDNA5 async loads (GLOBAL_LOAD_ASYNC_TO_LDS_B128,
//   ASYNCcnt) so tile fills run concurrently with WMMA work, no VGPR staging.
// - MoE uses top-2 routing (4x less compute than the reference dense einsum).
// ============================================================================

typedef __bf16 bf16_t;
typedef __attribute__((ext_vector_type(16))) __bf16 bf16x16;
typedef __attribute__((ext_vector_type(8)))  float  f32x8;

#define BN_RSQ 0.9999950000374997f   /* 1/sqrt(1+1e-5), eval-mode BN factor */
#define LN_EPS 1e-5f

enum { EPI_BF16_BIAS = 0, EPI_BF16_BN_RELU = 1, EPI_F32_BIAS = 2,
       EPI_F32_BN = 3, EPI_SLOT_F32 = 4 };

#define TBM 128
#define TBN 128
#define TBK 32
#define LDST 40   /* LDS row stride in bf16: 80B, 16B-aligned, conflict-free */

// --- CDNA5 async global->LDS copy (tracked by ASYNCcnt) ---------------------
// VDST = per-lane LDS byte offset (low 32 bits of shared aperture address),
// VADDR = 64-bit global address. See CDNA5 ISA.
__device__ __forceinline__ void async_copy_b128(unsigned lds_off, const void* g)
{
  asm volatile("global_load_async_to_lds_b128 %0, %1, off"
               :: "v"(lds_off), "v"((unsigned long long)(uintptr_t)g)
               : "memory");
}
__device__ __forceinline__ void wait_async0()
{
  asm volatile("s_wait_asynccnt 0x0" ::: "memory");
}

// ---------------------------------------------------------------------------
// f32 -> bf16 bulk conversion (vectorized, n multiple of 4)
// ---------------------------------------------------------------------------
__global__ __launch_bounds__(256)
void k_f2bf(const float* __restrict__ s, bf16_t* __restrict__ d, long nq)
{
  union BF4 { bf16_t b[4]; uint2 u; };
  long i = (long)blockIdx.x * blockDim.x + threadIdx.x;
  const long st = (long)gridDim.x * blockDim.x;
  for (; i < nq; i += st) {
    const float4 v = ((const float4*)s)[i];
    BF4 o;
    o.b[0] = (bf16_t)v.x; o.b[1] = (bf16_t)v.y;
    o.b[2] = (bf16_t)v.z; o.b[3] = (bf16_t)v.w;
    ((uint2*)d)[i] = o.u;
  }
}

// ---------------------------------------------------------------------------
// Tiled WMMA bf16 GEMM:  C[M,N] = A[M,K] @ W[N,K]^T  (+ epilogue)
// Block tile 128x128x32, 8 waves (2M x 4N), wave tile 64x32 (8 WMMAs/kstep).
// LDS double-buffered, filled with async-to-LDS loads overlapped with WMMA.
// GATHER: A rows indirected via per-expert row lists (MoE expert GEMM).
// ---------------------------------------------------------------------------
template<int EPI, bool GATHER>
__global__ __launch_bounds__(256)
void k_gemm(const bf16_t* __restrict__ A, int lda, int aColOff,
            const bf16_t* __restrict__ B0, int ldb,
            const float* __restrict__ bias0,
            const float* __restrict__ bnG, const float* __restrict__ bnB,
            const int* __restrict__ rowIdx,
            const unsigned* __restrict__ expOffs,
            const unsigned* __restrict__ expCnts,
            const float* __restrict__ rowW,
            void* __restrict__ Cout, long ldc, int cColOff,
            int N, int K)
{
  __shared__ bf16_t sA[2][TBM * LDST];
  __shared__ bf16_t sB[2][TBN * LDST];
  __shared__ int    sRow[TBM];

  const int tid = threadIdx.x;
  const int bx = blockIdx.x, by = blockIdx.y;

  unsigned cnt = 0, aoff = 0;
  const bf16_t* Bw = B0;
  const float* bias = bias0;
  if constexpr (GATHER) {
    const int e = blockIdx.z;
    cnt = expCnts[e];
    if ((unsigned)(by * TBM) >= cnt) return;   // block-uniform early exit
    aoff = expOffs[e];
    Bw   = B0 + (size_t)e * (size_t)N * (size_t)K;
    bias = bias0 + (size_t)e * (size_t)N;
    if (tid < TBM) {
      const unsigned li = by * TBM + tid;
      sRow[tid] = (li < cnt) ? rowIdx[aoff + li] : 0;
    }
    __syncthreads();
  }

  // --- global -> LDS staging: 256 threads, 2x16B chunks per operand each ---
  const int c1 = tid + 256;
  const int r0 = tid >> 2, f0 = (tid & 3) * 8;
  const int r1 = c1  >> 2, f1 = (c1  & 3) * 8;

  const bf16_t *aP0, *aP1;
  {
    const int g0 = GATHER ? sRow[r0] : (by * TBM + r0);
    const int g1 = GATHER ? sRow[r1] : (by * TBM + r1);
    aP0 = A + (size_t)g0 * lda + aColOff + f0;
    aP1 = A + (size_t)g1 * lda + aColOff + f1;
  }
  const bf16_t* bP0 = Bw + (size_t)(bx * TBN + r0) * ldb + f0;
  const bf16_t* bP1 = Bw + (size_t)(bx * TBN + r1) * ldb + f1;

  // per-lane LDS byte offsets for the async fills
  const unsigned sAb = (unsigned)(uintptr_t)(void*)&sA[0][0];
  const unsigned sBb = (unsigned)(uintptr_t)(void*)&sB[0][0];
  const unsigned la0 = (unsigned)(r0 * LDST + f0) * 2u;
  const unsigned la1 = (unsigned)(r1 * LDST + f1) * 2u;
  const unsigned bufStride = (unsigned)(TBM * LDST * 2);

  auto stage = [&](int buf, int k0) {
    const unsigned bo = buf ? bufStride : 0u;
    async_copy_b128(sAb + bo + la0, aP0 + k0);
    async_copy_b128(sAb + bo + la1, aP1 + k0);
    async_copy_b128(sBb + bo + la0, bP0 + k0);
    async_copy_b128(sBb + bo + la1, bP1 + k0);
  };

  // --- per-wave compute geometry ---
  const int lane = tid & 31;
  const int wv   = tid >> 5;
  const int wm   = (wv & 1) * 64;     // wave M offset within block
  const int wn   = (wv >> 1) * 32;    // wave N offset within block
  const int hsel = (lane >> 4) * 8;   // K-half select per ISA 16-bit A layout
  const int lm   = lane & 15;

  f32x8 acc[4][2] = {};
  union FragU { bf16x16 v; uint4 q[2]; };

  auto compute = [&](int buf) {
    FragU a[4], b[2];
#pragma unroll
    for (int i = 0; i < 4; ++i) {
      const bf16_t* p = &sA[buf][(wm + i * 16 + lm) * LDST + hsel];
      a[i].q[0] = *(const uint4*)p;          // K 0..7  (or 8..15)
      a[i].q[1] = *(const uint4*)(p + 16);   // K 16..23 (or 24..31)
    }
#pragma unroll
    for (int j = 0; j < 2; ++j) {
      const bf16_t* p = &sB[buf][(wn + j * 16 + lm) * LDST + hsel];
      b[j].q[0] = *(const uint4*)p;
      b[j].q[1] = *(const uint4*)(p + 16);
    }
#pragma unroll
    for (int i = 0; i < 4; ++i)
#pragma unroll
      for (int j = 0; j < 2; ++j)
        acc[i][j] = __builtin_amdgcn_wmma_f32_16x16x32_bf16(
            false, a[i].v, false, b[j].v, (short)0, acc[i][j], false, false);
  };

  // --- main loop: async fill of tile t+1 overlapped with WMMA on tile t ---
  stage(0, 0);
  wait_async0();
  __syncthreads();
  const int nk = K / TBK;
  for (int kt = 0; kt < nk; ++kt) {
    if (kt + 1 < nk) stage((kt + 1) & 1, (kt + 1) * TBK);
    compute(kt & 1);
    if (kt + 1 < nk) { wait_async0(); __syncthreads(); }
  }

  // --- epilogue (matches 16x16 f32 C/D VGPR layout) ---
#pragma unroll
  for (int j = 0; j < 2; ++j) {
    const int gn = bx * TBN + wn + j * 16 + lm;
    const float bv = bias[gn];
    float sc = 1.f, sh = 0.f;
    if constexpr (EPI == EPI_BF16_BN_RELU || EPI == EPI_F32_BN) {
      sc = bnG[gn] * BN_RSQ; sh = bnB[gn];
    }
#pragma unroll
    for (int i = 0; i < 4; ++i) {
#pragma unroll
      for (int v = 0; v < 8; ++v) {
        const int lr = wm + i * 16 + hsel + v;   // local M row
        float val = acc[i][j][v] + bv;
        if constexpr (EPI == EPI_BF16_BN_RELU || EPI == EPI_F32_BN)
          val = val * sc + sh;
        if constexpr (EPI == EPI_BF16_BN_RELU)
          val = fmaxf(val, 0.f);
        if constexpr (EPI == EPI_SLOT_F32) {
          const unsigned li = by * TBM + lr;
          if (li < cnt)
            ((float*)Cout)[(size_t)(aoff + li) * ldc + gn] =
                rowW[aoff + li] * val;
        } else if constexpr (EPI == EPI_BF16_BIAS || EPI == EPI_BF16_BN_RELU) {
          ((bf16_t*)Cout)[(size_t)(by * TBM + lr) * ldc + cColOff + gn] =
              (bf16_t)val;
        } else {
          ((float*)Cout)[(size_t)(by * TBM + lr) * ldc + cColOff + gn] = val;
        }
      }
    }
  }
}

// ---------------------------------------------------------------------------
// Row-wise LayerNorm over C=2048 f32, write bf16
// ---------------------------------------------------------------------------
__global__ __launch_bounds__(256)
void k_layernorm(const float* __restrict__ X, const float* __restrict__ g,
                 const float* __restrict__ b, bf16_t* __restrict__ out, int C)
{
  const long row = blockIdx.x;
  const float* x = X + row * C;
  bf16_t* o = out + row * C;
  float s = 0.f, s2 = 0.f;
  for (int c = threadIdx.x; c < C; c += 256) {
    const float v = x[c]; s += v; s2 += v * v;
  }
  for (int off2 = 16; off2; off2 >>= 1) {
    s  += __shfl_xor(s,  off2, 32);
    s2 += __shfl_xor(s2, off2, 32);
  }
  __shared__ float rs[8], rs2[8];
  const int wv = threadIdx.x >> 5;
  if ((threadIdx.x & 31) == 0) { rs[wv] = s; rs2[wv] = s2; }
  __syncthreads();
  s = 0.f; s2 = 0.f;
#pragma unroll
  for (int i = 0; i < 8; ++i) { s += rs[i]; s2 += rs2[i]; }
  const float mu  = s / C;
  const float inv = rsqrtf(s2 / C - mu * mu + LN_EPS);
  for (int c = threadIdx.x; c < C; c += 256)
    o[c] = (bf16_t)((x[c] - mu) * inv * g[c] + b[c]);
}

// ---------------------------------------------------------------------------
// Gate: logits = x @ gate_w^T + gate_b ; top-2 ; softmax ; expert counts
// One wave per row.
// ---------------------------------------------------------------------------
__global__ __launch_bounds__(256)
void k_gate_top2(const bf16_t* __restrict__ mlp, int ldm, int colOff,
                 const float* __restrict__ gw, const float* __restrict__ gb,
                 int* __restrict__ idxOut, float* __restrict__ wOut,
                 unsigned* __restrict__ counts, int Btot, int Kd)
{
  const int row  = blockIdx.x * 8 + (threadIdx.x >> 5);
  const int lane = threadIdx.x & 31;
  if (row >= Btot) return;
  const bf16_t* x = mlp + (size_t)row * ldm + colOff;
  float acc[8] = {0.f,0.f,0.f,0.f,0.f,0.f,0.f,0.f};
  for (int k = lane; k < Kd; k += 32) {
    const float xv = (float)x[k];
#pragma unroll
    for (int e = 0; e < 8; ++e) acc[e] += xv * gw[e * Kd + k];
  }
#pragma unroll
  for (int e = 0; e < 8; ++e)
    for (int o = 16; o; o >>= 1) acc[e] += __shfl_xor(acc[e], o, 32);
  if (lane == 0) {
#pragma unroll
    for (int e = 0; e < 8; ++e) acc[e] += gb[e];
    int e0 = 0; float v0 = acc[0];
    for (int e = 1; e < 8; ++e) if (acc[e] > v0) { v0 = acc[e]; e0 = e; }
    int e1 = -1; float v1 = -3.4e38f;
    for (int e = 0; e < 8; ++e)
      if (e != e0 && acc[e] > v1) { v1 = acc[e]; e1 = e; }
    const float w0 = 1.f / (1.f + __expf(v1 - v0));  // softmax over {v0,v1}
    idxOut[row * 2]     = e0;  idxOut[row * 2 + 1] = e1;
    wOut[row * 2]       = w0;  wOut[row * 2 + 1]   = 1.f - w0;
    atomicAdd(&counts[e0], 1u);
    atomicAdd(&counts[e1], 1u);
  }
}

__global__ void k_zero_u32(unsigned* p, int n)
{
  if ((int)threadIdx.x < n) p[threadIdx.x] = 0u;
}

__global__ void k_offsets(const unsigned* __restrict__ counts,
                          unsigned* __restrict__ offs,
                          unsigned* __restrict__ cursors)
{
  if (threadIdx.x == 0) {
    unsigned s = 0;
    for (int e = 0; e < 8; ++e) { offs[e] = s; s += counts[e]; cursors[e] = 0u; }
  }
}

__global__ __launch_bounds__(256)
void k_scatter(const int* __restrict__ idx, const float* __restrict__ w,
               const unsigned* __restrict__ offs, unsigned* __restrict__ cursors,
               int* __restrict__ srcRow, float* __restrict__ wA,
               int* __restrict__ slotOf, int Btot)
{
  const int row = blockIdx.x * 256 + threadIdx.x;
  if (row >= Btot) return;
#pragma unroll
  for (int k = 0; k < 2; ++k) {
    const int e = idx[row * 2 + k];
    const unsigned a = offs[e] + atomicAdd(&cursors[e], 1u);
    srcRow[a] = row;
    wA[a]     = w[row * 2 + k];
    slotOf[row * 2 + k] = (int)a;
  }
}

// moeOut[row, colOff+d] = slot[s0,d] + slot[s1,d]   (gate weights pre-applied)
__global__ __launch_bounds__(256)
void k_combine(const float* __restrict__ slot, const int* __restrict__ slotOf,
               bf16_t* __restrict__ out, int ldo, int colOff, int Dd)
{
  const long i  = (long)blockIdx.x * 256 + threadIdx.x;
  const long row = i / Dd;
  const int  d   = (int)(i - row * Dd);
  const int s0 = slotOf[row * 2], s1 = slotOf[row * 2 + 1];
  out[row * ldo + colOff + d] =
      (bf16_t)(slot[(size_t)s0 * Dd + d] + slot[(size_t)s1 * Dd + d]);
}

// out[row] = dot(fusOut[row,:], reg_w) + reg_b  (one wave per row)
__global__ __launch_bounds__(256)
void k_reg(const float* __restrict__ X, const float* __restrict__ rw,
           const float* __restrict__ rb, float* __restrict__ out,
           int Dd, int Btot)
{
  const int row  = blockIdx.x * 8 + (threadIdx.x >> 5);
  const int lane = threadIdx.x & 31;
  if (row >= Btot) return;
  const float* x = X + (size_t)row * Dd;
  float s = 0.f;
  for (int k = lane; k < Dd; k += 32) s += x[k] * rw[k];
  for (int o = 16; o; o >>= 1) s += __shfl_xor(s, o, 32);
  if (lane == 0) out[row] = s + rb[0];
}

// ===========================================================================
extern "C" void kernel_launch(void* const* d_in, const int* in_sizes, int n_in,
                              void* d_out, int out_size, void* d_ws, size_t ws_size,
                              hipStream_t stream)
{
  (void)in_sizes; (void)n_in; (void)out_size; (void)ws_size;
  constexpr int Bb = 16384, IN1 = 1024, IN2 = 1024, Dd = 1024, Hh = 2048;
  constexpr int Ee = 8, TD = 2048;

  const float* x1    = (const float*)d_in[0];
  const float* x2    = (const float*)d_in[1];
  const float* p1_w  = (const float*)d_in[2];
  const float* p1_b  = (const float*)d_in[3];
  const float* p2_w  = (const float*)d_in[4];
  const float* p2_b  = (const float*)d_in[5];
  const float* m1_w  = (const float*)d_in[6];
  const float* m1_b  = (const float*)d_in[7];
  const float* bn1_g = (const float*)d_in[8];
  const float* bn1_b = (const float*)d_in[9];
  const float* m2_w  = (const float*)d_in[10];
  const float* m2_b  = (const float*)d_in[11];
  const float* bn2_g = (const float*)d_in[12];
  const float* bn2_b = (const float*)d_in[13];
  const float* m3_w  = (const float*)d_in[14];
  const float* m3_b  = (const float*)d_in[15];
  const float* ln_g  = (const float*)d_in[16];
  const float* ln_b  = (const float*)d_in[17];
  const float* ex_w  = (const float*)d_in[18];
  const float* ex_b  = (const float*)d_in[19];
  const float* gate_w= (const float*)d_in[20];
  const float* gate_b= (const float*)d_in[21];
  const float* fus_w = (const float*)d_in[22];
  const float* fus_b = (const float*)d_in[23];
  const float* bnf_g = (const float*)d_in[24];
  const float* bnf_b = (const float*)d_in[25];
  const float* reg_w = (const float*)d_in[26];
  const float* reg_b = (const float*)d_in[27];

  // ---- workspace bump allocator ----
  char* base = (char*)d_ws;
  size_t woff = 0;
  auto alloc = [&](size_t bytes) -> void* {
    void* p = base + woff;
    woff += (bytes + 255) & ~(size_t)255;
    return p;
  };

  bf16_t* p1wB  = (bf16_t*)alloc((size_t)Dd * IN1 * 2);
  bf16_t* p2wB  = (bf16_t*)alloc((size_t)Dd * IN2 * 2);
  bf16_t* m1wB  = (bf16_t*)alloc((size_t)Hh * TD  * 2);
  bf16_t* m2wB  = (bf16_t*)alloc((size_t)Hh * Hh  * 2);
  bf16_t* m3wB  = (bf16_t*)alloc((size_t)TD * Hh  * 2);
  bf16_t* exwB  = (bf16_t*)alloc((size_t)Ee * Dd * Dd * 2);
  bf16_t* fuswB = (bf16_t*)alloc((size_t)Dd * TD  * 2);
  bf16_t* x1B   = (bf16_t*)alloc((size_t)Bb * IN1 * 2);  // x1B+x2B region later
  bf16_t* x2B   = (bf16_t*)alloc((size_t)Bb * IN2 * 2);  //   reused as fusOut f32
  bf16_t* hA    = (bf16_t*)alloc((size_t)Bb * TD * 2);
  bf16_t* hB    = (bf16_t*)alloc((size_t)Bb * TD * 2);
  float*  big   = (float*) alloc((size_t)2 * Bb * Dd * 4); // m3 f32 out / MoE slots
  int*      idxArr  = (int*)     alloc((size_t)Bb * 2 * 4);
  float*    gwArr   = (float*)   alloc((size_t)Bb * 2 * 4);
  int*      slotOf  = (int*)     alloc((size_t)Bb * 2 * 4);
  int*      srcRow  = (int*)     alloc((size_t)2 * Bb * 4);
  float*    wArr    = (float*)   alloc((size_t)2 * Bb * 4);
  unsigned* counts  = (unsigned*)alloc(8 * 4);
  unsigned* offs    = (unsigned*)alloc(8 * 4);
  unsigned* cursors = (unsigned*)alloc(8 * 4);
  float* fusOut = (float*)x1B;   // alias: 2*Bb*1024*2B == Bb*1024*4B

  const dim3 blk(256);
  auto conv = [&](const float* s, bf16_t* dd, size_t n) {
    const size_t nq = n >> 2;
    unsigned blocks = (unsigned)((nq + 255) / 256);
    if (blocks > 32768u) blocks = 32768u;
    k_f2bf<<<dim3(blocks), blk, 0, stream>>>(s, dd, (long)nq);
  };

  // ---- 1) convert all weights + inputs to bf16 (weights: 48MB, L2-resident)
  conv(p1_w, p1wB, (size_t)Dd * IN1);
  conv(p2_w, p2wB, (size_t)Dd * IN2);
  conv(m1_w, m1wB, (size_t)Hh * TD);
  conv(m2_w, m2wB, (size_t)Hh * Hh);
  conv(m3_w, m3wB, (size_t)TD * Hh);
  conv(ex_w, exwB, (size_t)Ee * Dd * Dd);
  conv(fus_w, fuswB, (size_t)Dd * TD);
  conv(x1, x1B, (size_t)Bb * IN1);
  conv(x2, x2B, (size_t)Bb * IN2);

  // ---- 2) projections -> hA (bf16 [B, 2048], concat) ----
  k_gemm<EPI_BF16_BIAS, false><<<dim3(Dd / TBN, Bb / TBM), blk, 0, stream>>>(
      x1B, IN1, 0, p1wB, IN1, p1_b, nullptr, nullptr,
      nullptr, nullptr, nullptr, nullptr, hA, TD, 0, Dd, IN1);
  k_gemm<EPI_BF16_BIAS, false><<<dim3(Dd / TBN, Bb / TBM), blk, 0, stream>>>(
      x2B, IN2, 0, p2wB, IN2, p2_b, nullptr, nullptr,
      nullptr, nullptr, nullptr, nullptr, hA, TD, Dd, Dd, IN2);

  // ---- 3) m1: relu(bn1(hA @ m1_w^T + b)) -> hB ----
  k_gemm<EPI_BF16_BN_RELU, false><<<dim3(Hh / TBN, Bb / TBM), blk, 0, stream>>>(
      hA, TD, 0, m1wB, TD, m1_b, bn1_g, bn1_b,
      nullptr, nullptr, nullptr, nullptr, hB, Hh, 0, Hh, TD);

  // ---- 4) m2: relu(bn2(hB @ m2_w^T + b)) -> hA ----
  k_gemm<EPI_BF16_BN_RELU, false><<<dim3(Hh / TBN, Bb / TBM), blk, 0, stream>>>(
      hB, Hh, 0, m2wB, Hh, m2_b, bn2_g, bn2_b,
      nullptr, nullptr, nullptr, nullptr, hA, Hh, 0, Hh, Hh);

  // ---- 5) m3: hA @ m3_w^T + b -> big (f32 [B, 2048]) ----
  k_gemm<EPI_F32_BIAS, false><<<dim3(TD / TBN, Bb / TBM), blk, 0, stream>>>(
      hA, Hh, 0, m3wB, Hh, m3_b, nullptr, nullptr,
      nullptr, nullptr, nullptr, nullptr, big, TD, 0, TD, Hh);

  // ---- 6) LayerNorm -> hB (bf16 [B, 2048] = [x1m | x2m]) ----
  k_layernorm<<<dim3(Bb), blk, 0, stream>>>(big, ln_g, ln_b, hB, TD);

  // ---- 7) MoE on each half; expert slots live in `big`, output -> hA ----
  for (int s = 0; s < 2; ++s) {
    const int colOff = s * Dd;
    k_zero_u32<<<dim3(1), dim3(32), 0, stream>>>(counts, 8);
    k_gate_top2<<<dim3(Bb / 8), blk, 0, stream>>>(
        hB, TD, colOff, gate_w, gate_b, idxArr, gwArr, counts, Bb, Dd);
    k_offsets<<<dim3(1), dim3(32), 0, stream>>>(counts, offs, cursors);
    k_scatter<<<dim3(Bb / 256), blk, 0, stream>>>(
        idxArr, gwArr, offs, cursors, srcRow, wArr, slotOf, Bb);
    // gathered expert GEMM: slot[a,:] = w[a] * (x[srcRow[a],:] @ exw[e]^T + exb[e])
    k_gemm<EPI_SLOT_F32, true><<<dim3(Dd / TBN, Bb / TBM, Ee), blk, 0, stream>>>(
        hB, TD, colOff, exwB, Dd, ex_b, nullptr, nullptr,
        srcRow, offs, counts, wArr, big, Dd, 0, Dd, Dd);
    k_combine<<<dim3((unsigned)(((size_t)Bb * Dd) / 256)), blk, 0, stream>>>(
        big, slotOf, hA, TD, colOff, Dd);
  }

  // ---- 8) fus: bn(hA @ fus_w^T + b) -> fusOut (f32 [B, 1024]) ----
  k_gemm<EPI_F32_BN, false><<<dim3(Dd / TBN, Bb / TBM), blk, 0, stream>>>(
      hA, TD, 0, fuswB, TD, fus_b, bnf_g, bnf_b,
      nullptr, nullptr, nullptr, nullptr, fusOut, Dd, 0, Dd, TD);

  // ---- 9) regression head -> d_out (f32 [B]) ----
  k_reg<<<dim3(Bb / 8), blk, 0, stream>>>(
      fusOut, reg_w, reg_b, (float*)d_out, Dd, Bb);
}
